// FragmentedLinear_64089501991434
// MI455X (gfx1250) — matrix operand order repaired and checked by reading, not compiled, not run on hardware
//
#include <hip/hip_runtime.h>
#include <hip/hip_bf16.h>

// ---------------------------------------------------------------------------
// Problem constants: B=8192, IN=4096, OUT=4096, F=16, COMP=1024, FS=256.
// ---------------------------------------------------------------------------
static constexpr int Bsz  = 8192;
static constexpr int INF  = 4096;
static constexpr int OUTF = 4096;
static constexpr int NF   = 16;
static constexpr int COMP = 1024;
static constexpr int FS   = 256;

typedef __bf16 bf16_t;
typedef __attribute__((ext_vector_type(16))) __bf16 v16bf;
typedef __attribute__((ext_vector_type(8)))  __bf16 v8bf;
typedef __attribute__((ext_vector_type(8)))  float  v8f;

// ---------------------------------------------------------------------------
// CDNA5 async global->LDS copy (ASYNCcnt-tracked). LDS address = low 32 bits
// of the generic pointer (LDS aperture: addr[31:0] is the LDS offset).
// ---------------------------------------------------------------------------
static __device__ inline void async_copy16(uint32_t lds_addr, const bf16_t* gptr) {
    asm volatile("global_load_async_to_lds_b128 %0, %1, off"
                 :: "v"(lds_addr), "v"(gptr)
                 : "memory");
}
static __device__ inline void wait_async_le8() {
    asm volatile("s_wait_asynccnt 0x8" ::: "memory");
}
static __device__ inline void wait_async_0() {
    asm volatile("s_wait_asynccnt 0x0" ::: "memory");
}

// ---------------------------------------------------------------------------
// Kernel 1: fragment scores + softmax -> probs [B, 16]
// ---------------------------------------------------------------------------
__global__ __launch_bounds__(256)
void frag_softmax_kernel(const float* __restrict__ x,
                         const float* __restrict__ selw,
                         float* __restrict__ probs) {
    const int row = blockIdx.x;
    const int t   = threadIdx.x;       // 0..255
    const int f   = t >> 4;            // fragment 0..15
    const int s   = t & 15;            // 16 threads per fragment

    const float* xr = x    + (size_t)row * INF + f * FS + s * 16;
    const float* wr = selw + f * FS + s * 16;
    float acc = 0.f;
#pragma unroll
    for (int i = 0; i < 16; i += 4) {
        float4 xv = *(const float4*)(xr + i);
        float4 wv = *(const float4*)(wr + i);
        acc += xv.x * wv.x + xv.y * wv.y + xv.z * wv.z + xv.w * wv.w;
    }
#pragma unroll
    for (int off = 8; off >= 1; off >>= 1)
        acc += __shfl_xor(acc, off, 16);

    __shared__ float sc[NF];
    if (s == 0) sc[f] = acc;
    __syncthreads();

    if (t < NF) {
        float m = -1e30f;
#pragma unroll
        for (int i = 0; i < NF; ++i) m = fmaxf(m, sc[i]);
        float ssum = 0.f;
#pragma unroll
        for (int i = 0; i < NF; ++i) ssum += __expf(sc[i] - m);
        probs[(size_t)row * NF + t] = __expf(sc[t] - m) / ssum;
    }
}

// ---------------------------------------------------------------------------
// Kernel 2: gated split. Xg = bf16(x*p), Xm = bf16(x - x*p). One pass over x.
// ---------------------------------------------------------------------------
__global__ __launch_bounds__(256)
void gate_split_kernel(const float* __restrict__ x,
                       const float* __restrict__ probs,
                       bf16_t* __restrict__ xg,
                       bf16_t* __restrict__ xm) {
    size_t idx = (size_t)blockIdx.x * blockDim.x + threadIdx.x; // float4 id
    size_t e   = idx * 4;
    int row  = (int)(e >> 12);            // / IN (4096)
    int col  = (int)(e & (INF - 1));
    float p  = probs[(size_t)row * NF + (col >> 8)];   // FS = 256
    float4 xv = *(const float4*)(x + e);
    float g0 = xv.x * p, g1 = xv.y * p, g2 = xv.z * p, g3 = xv.w * p;
    xg[e + 0] = (bf16_t)g0;           xm[e + 0] = (bf16_t)(xv.x - g0);
    xg[e + 1] = (bf16_t)g1;           xm[e + 1] = (bf16_t)(xv.y - g1);
    xg[e + 2] = (bf16_t)g2;           xm[e + 2] = (bf16_t)(xv.z - g2);
    xg[e + 3] = (bf16_t)g3;           xm[e + 3] = (bf16_t)(xv.w - g3);
}

// ---------------------------------------------------------------------------
// Kernel 3: flat fp32 -> bf16 convert (W_comp / W_net are already N-major).
// ---------------------------------------------------------------------------
__global__ __launch_bounds__(256)
void cvt_bf16_kernel(const float* __restrict__ in, bf16_t* __restrict__ out) {
    size_t idx = (size_t)blockIdx.x * blockDim.x + threadIdx.x;
    size_t e   = idx * 4;
    float4 v = *(const float4*)(in + e);
    out[e + 0] = (bf16_t)v.x;
    out[e + 1] = (bf16_t)v.y;
    out[e + 2] = (bf16_t)v.z;
    out[e + 3] = (bf16_t)v.w;
}

// ---------------------------------------------------------------------------
// Kernel 4: transpose + convert: in [R][C] fp32 -> out [C][R] bf16.
// (Expert weights only: [IN,OUT] -> [OUT,IN] N-major.)
// ---------------------------------------------------------------------------
__global__ __launch_bounds__(256)
void transpose_cvt_kernel(const float* __restrict__ in, bf16_t* __restrict__ out,
                          int R, int C) {
    __shared__ float tile[32][33];
    int c0 = blockIdx.x * 32, r0 = blockIdx.y * 32;
    int tx = threadIdx.x, ty = threadIdx.y;
#pragma unroll
    for (int i = 0; i < 32; i += 8)
        tile[ty + i][tx] = in[(size_t)(r0 + ty + i) * C + (c0 + tx)];
    __syncthreads();
#pragma unroll
    for (int i = 0; i < 32; i += 8)
        out[(size_t)(c0 + ty + i) * R + (r0 + tx)] = (bf16_t)tile[tx][ty + i];
}

// ---------------------------------------------------------------------------
// Kernel 5: bf16 WMMA GEMM, async-copy double-buffered, BK=64.
// C[M,N] = A[M,K] @ Bt[N,K]^T. MODE 0: fp32 store. MODE 1: bf16 store.
// MODE 2: fp32 store of acc + Cadd.
//
// Block tile 128x128, K-step 64 split into two register-reusing 32-halves
// (16 WMMAs per barrier interval). 256 threads = 8 waves (2 M x 4 N); each
// wave owns 64x32 = 4x2 accumulators of 16x16. Staging uses
// global_load_async_to_lds_b128, 2-deep buffered (step k+1 in flight while
// computing step k; s_wait_asynccnt 8 retains next step's 8 ops). LDS pitch
// 72 bf16 (144B stride) is bank-conflict-free for b128 fragment loads.
// All dims are exact tile multiples: EXEC all-ones around every WMMA.
// ---------------------------------------------------------------------------
#define BM 128
#define BN 128
#define BK 64
#define LDT 72                       // LDS row pitch in bf16 elems (64+8 pad)
#define BUFE ((BM + BN) * LDT)       // bf16 elems per buffer (A tile + B tile)

static __device__ inline v16bf combine8(v8bf lo, v8bf hi) {
    v16bf r;
#pragma unroll
    for (int i = 0; i < 8; ++i) { r[i] = lo[i]; r[i + 8] = hi[i]; }
    return r;
}

template <int MODE>
__global__ __launch_bounds__(256)
void wmma_gemm_bf16_kernel(const bf16_t* __restrict__ A,   // [M][K]
                           const bf16_t* __restrict__ Bt,  // [N][K]
                           const float* __restrict__ Cadd, // MODE 2 only
                           void* __restrict__ Cout,
                           int M, int N, int K) {
    __shared__ bf16_t smem[2 * BUFE];   // buf0 | buf1; each: As[128][72], Bs[128][72]

    const int tid  = threadIdx.x;
    const int lane = tid & 31;
    const int wave = tid >> 5;
    const int bm = blockIdx.y * BM;
    const int bn = blockIdx.x * BN;
    const int waveM = (wave & 1) * 64;
    const int waveN = (wave >> 1) * 32;
    const int l16   = lane & 15;
    const int aHalf = (lane < 16) ? 0 : 8;    // ISA 16-bit A layout K-half
    const int bHalf = (lane < 16) ? 0 : 16;   // unused; B fragments use aHalf (same K order as A)

    // staging geometry: 1024 x 16B chunks per operand, 4 per thread
    // chunk id = tid + c*256: row = id>>3 (8 chunks per 64-elem row), kk = (id&7)*8
    const int rs = tid >> 3;                  // base row 0..31 (+32 per c)
    const int kk = (tid & 7) * 8;             // 0..56

    // per-thread LDS byte addresses (buffer 0); buffer 1 = +2*BUFE bytes
    uint32_t ldsA[4], ldsB[4];
    const bf16_t* gA[4];
    const bf16_t* gB[4];
#pragma unroll
    for (int c = 0; c < 4; ++c) {
        int r = rs + c * 32;
        ldsA[c] = (uint32_t)(uintptr_t)&smem[r * LDT + kk];
        ldsB[c] = (uint32_t)(uintptr_t)&smem[BM * LDT + r * LDT + kk];
        gA[c]   = A  + (size_t)(bm + r) * K + kk;
        gB[c]   = Bt + (size_t)(bn + r) * K + kk;
    }
    const uint32_t bufStride = 2u * BUFE;     // bytes per buffer

    auto issue = [&](int k0, int buf) {
        uint32_t bo = buf ? bufStride : 0u;
#pragma unroll
        for (int c = 0; c < 4; ++c) async_copy16(ldsA[c] + bo, gA[c] + k0);
#pragma unroll
        for (int c = 0; c < 4; ++c) async_copy16(ldsB[c] + bo, gB[c] + k0);
    };

    v8f acc[4][2] = {};

    // prologue: fill the 2-deep pipeline
    issue(0, 0);
    if (BK < K) issue(BK, 1);

    int buf = 0;
    for (int k0 = 0; k0 < K; k0 += BK, buf ^= 1) {
        if (k0 + BK < K) wait_async_le8();    // step k done, step k+1 in flight
        else             wait_async_0();      // last step: drain
        __syncthreads();                      // all waves see the full buffer

        const bf16_t* Asb = smem + (buf ? BUFE : 0);
        const bf16_t* Bsb = Asb + BM * LDT;

        // two 32-wide K-halves reuse the same fragment registers
#pragma unroll
        for (int h = 0; h < 2; ++h) {
            const int hk = h * 32;
            v16bf afrag[4], bfrag[2];
#pragma unroll
            for (int i = 0; i < 4; ++i) {
                const bf16_t* ap = Asb + (waveM + i * 16 + l16) * LDT + hk;
                afrag[i] = combine8(*(const v8bf*)(ap + aHalf),
                                    *(const v8bf*)(ap + 16 + aHalf));
            }
#pragma unroll
            for (int j = 0; j < 2; ++j) {
                const bf16_t* bp = Bsb + (waveN + j * 16 + l16) * LDT + hk;
                bfrag[j] = combine8(*(const v8bf*)(bp + aHalf),
                                    *(const v8bf*)(bp + 16 + aHalf));
            }
#pragma unroll
            for (int i = 0; i < 4; ++i)
#pragma unroll
                for (int j = 0; j < 2; ++j)
                    acc[i][j] = __builtin_amdgcn_wmma_f32_16x16x32_bf16(
                        false, afrag[i], false, bfrag[j],
                        (short)0, acc[i][j], false, false);
        }

        __syncthreads();                      // all waves done reading buf
        if (k0 + 2 * BK < K) issue(k0 + 2 * BK, buf);  // refill freed buffer
    }

    // --- straight-line epilogue ---------------------------------------------
    const int rofs = (lane < 16) ? 0 : 8;
    float*  Cf = (float*)Cout;
    bf16_t* Cb = (bf16_t*)Cout;
#pragma unroll
    for (int i = 0; i < 4; ++i) {
#pragma unroll
        for (int j = 0; j < 2; ++j) {
            int col = bn + waveN + j * 16 + l16;
#pragma unroll
            for (int r = 0; r < 8; ++r) {
                int row = bm + waveM + i * 16 + rofs + r;
                size_t o = (size_t)row * N + col;
                float v = acc[i][j][r];
                if constexpr (MODE == 2) v += Cadd[o];
                if constexpr (MODE == 1) Cb[o] = (bf16_t)v;
                else                     Cf[o] = v;
            }
        }
    }
}

// ---------------------------------------------------------------------------
// Host launch
// ---------------------------------------------------------------------------
extern "C" void kernel_launch(void* const* d_in, const int* in_sizes, int n_in,
                              void* d_out, int out_size, void* d_ws, size_t ws_size,
                              hipStream_t stream) {
    const float* x     = (const float*)d_in[0];   // [B, IN]
    const float* selw  = (const float*)d_in[1];   // [F, FS]
    const float* expw  = (const float*)d_in[2];   // [F, FS, OUT] == [IN, OUT]
    const float* wcomp = (const float*)d_in[3];   // [COMP, IN]   (N-major already)
    const float* wnet  = (const float*)d_in[4];   // [OUT, COMP]  (N-major already)
    float* out = (float*)d_out;                   // [B, OUT] fp32

    char*  ws  = (char*)d_ws;
    size_t off = 0;
    auto carve = [&](size_t bytes) -> void* {
        off = (off + 255) & ~(size_t)255;
        void* p = ws + off;
        off += bytes;
        return p;
    };
    float*  probs  = (float*) carve((size_t)Bsz * NF * sizeof(float));
    bf16_t* xg     = (bf16_t*)carve((size_t)Bsz * INF * sizeof(bf16_t));
    bf16_t* xm     = (bf16_t*)carve((size_t)Bsz * INF * sizeof(bf16_t));
    bf16_t* webf   = (bf16_t*)carve((size_t)OUTF * INF * sizeof(bf16_t));  // [OUT][IN]
    bf16_t* wcbf   = (bf16_t*)carve((size_t)COMP * INF * sizeof(bf16_t));  // [COMP][IN]
    bf16_t* wnbf   = (bf16_t*)carve((size_t)OUTF * COMP * sizeof(bf16_t)); // [OUT][COMP]
    bf16_t* compbf = (bf16_t*)carve((size_t)Bsz * COMP * sizeof(bf16_t));

    // 1) fragment scores + softmax
    frag_softmax_kernel<<<Bsz, 256, 0, stream>>>(x, selw, probs);

    // 2) gated split (x -> Xg, Xm bf16)
    gate_split_kernel<<<(unsigned)((size_t)Bsz * INF / 4 / 256), 256, 0, stream>>>(
        x, probs, xg, xm);

    // 3) weight prep
    transpose_cvt_kernel<<<dim3(OUTF / 32, INF / 32), dim3(32, 8), 0, stream>>>(
        expw, webf, INF, OUTF);   // [IN,OUT] -> [OUT,IN]
    cvt_bf16_kernel<<<(unsigned)((size_t)COMP * INF / 4 / 256), 256, 0, stream>>>(
        wcomp, wcbf);
    cvt_bf16_kernel<<<(unsigned)((size_t)OUTF * COMP / 4 / 256), 256, 0, stream>>>(
        wnet, wnbf);

    // 4) GEMM 1: expert_out = Xg @ We  -> d_out (fp32)
    wmma_gemm_bf16_kernel<0><<<dim3(OUTF / BN, Bsz / BM), 256, 0, stream>>>(
        xg, webf, nullptr, out, Bsz, OUTF, INF);

    // 5) GEMM 2: compressed = Xm @ Wc^T -> bf16 workspace
    wmma_gemm_bf16_kernel<1><<<dim3(COMP / BN, Bsz / BM), 256, 0, stream>>>(
        xm, wcbf, nullptr, compbf, Bsz, COMP, INF);

    // 6) GEMM 3: out = compressed @ Wn^T + expert_out (in-place add)
    wmma_gemm_bf16_kernel<2><<<dim3(OUTF / BN, Bsz / BM), 256, 0, stream>>>(
        compbf, wnbf, out, out, Bsz, OUTF, COMP);
}
